// SparseConvolutionBase_37271726195534
// MI455X (gfx1250) — compile-verified, matrix operand order checked by reading, not run on hardware
//
#include <hip/hip_runtime.h>
#include <hip/hip_bf16.h>

// CDNA5 / gfx1250 sparse convolution: gather -> bf16 WMMA GEMM -> atomic scatter.
//
// Roofline: 44.2 GFLOP, ~345 MB bf16 gather + ~690 MB fp32 atomic scatter, all
// L2-resident (51 MB features + 51 MB out << 192 MB L2).  bf16 WMMA with f32
// accumulation keeps numerics near the fp32 reference while using the matrix
// pipe; fp32->bf16 conversion is done once in workspace to halve gather bytes.
//
// v4: B fragments for a whole nt-tile preloaded into registers before the
// WMMA chain so the 8 ds_load_b128 issue as one clause and drain with partial
// dscnt waits (no per-WMMA s_wait_dscnt 0x0 serialization).  Scatter stays
// saddr-form 32-bit-offset atomics with the nt column in the immediate.

typedef __attribute__((ext_vector_type(16))) __bf16 v16bf;
typedef __attribute__((ext_vector_type(8)))  float  v8f;

static __device__ __forceinline__ unsigned short f32_to_bf16_rne(float f) {
  union { float f; unsigned u; } v; v.f = f;
  unsigned u = v.u;
  unsigned r = u + 0x7FFFu + ((u >> 16) & 1u);   // round-to-nearest-even
  return (unsigned short)(r >> 16);
}

// ---- prep kernel 1: weights fp32 [k][c][d] -> bf16 transposed [k][d][c] ----
__global__ void spc_conv_weights(const float* __restrict__ w,
                                 unsigned short* __restrict__ wt, int total) {
  int idx = blockIdx.x * blockDim.x + threadIdx.x;
  if (idx >= total) return;
  int k   = idx >> 14;          // 128*128 = 16384 per offset
  int rem = idx & 16383;
  int c   = rem >> 7;
  int d   = rem & 127;
  wt[((size_t)k << 14) + (d << 7) + c] = f32_to_bf16_rne(w[idx]);
}

// ---- prep kernel 2: features fp32 -> bf16, same layout (vectorized x4) ----
__global__ void spc_conv_feat(const float4* __restrict__ src,
                              uint2* __restrict__ dst, int n4) {
  int i = blockIdx.x * blockDim.x + threadIdx.x;
  if (i >= n4) return;
  float4 v = src[i];
  uint2 o;
  o.x = (unsigned)f32_to_bf16_rne(v.x) | ((unsigned)f32_to_bf16_rne(v.y) << 16);
  o.y = (unsigned)f32_to_bf16_rne(v.z) | ((unsigned)f32_to_bf16_rne(v.w) << 16);
  dst[i] = o;
}

// ---- prep kernel 3: out[r][c] = bias[c] (vectorized x4) ----
__global__ void spc_bias_init(const float4* __restrict__ bias4,
                              float4* __restrict__ out4, int n4) {
  int i = blockIdx.x * blockDim.x + threadIdx.x;
  if (i >= n4) return;
  out4[i] = bias4[i & 31];      // 128 floats / 4 = 32 float4 of bias
}

// ---- main kernel: per (k, m-tile) gather -> WMMA -> atomic scatter ----
#define LDS_PITCH 136           // 128 + 8 halves; 272B row stride spreads banks

__global__ __launch_bounds__(256)
void spc_gemm(const unsigned short* __restrict__ featbf,   // [N_in][128] bf16
              const unsigned short* __restrict__ wtbf,     // [K][d=128][c=128] bf16
              const int* __restrict__ in_map,              // [K][M]
              const int* __restrict__ out_map,             // [K][M]
              float* __restrict__ out,                     // [N_out][128]
              int Mpairs) {
  __shared__ unsigned short ldsW[128 * LDS_PITCH];

  const int k    = blockIdx.y;
  const int m0   = blockIdx.x * 128;
  const int tid  = threadIdx.x;
  const int lane = tid & 31;
  const int wave = tid >> 5;
  const int Mlast = Mpairs - 1;
  const bool fullTile = (m0 + 128) <= Mpairs;   // uniform across block

  // Stage W^T[k] (32 KB bf16) into LDS with padded pitch.
  {
    const uint4* src = (const uint4*)(wtbf + ((size_t)k << 14));
#pragma unroll
    for (int i = 0; i < 8; ++i) {
      int idx = tid + i * 256;            // 2048 x uint4 = 32 KB
      int d   = idx >> 4;                 // 16 uint4 per 128-half row
      int ch  = idx & 15;
      uint4 v = src[idx];
      *(uint4*)&ldsW[d * LDS_PITCH + ch * 8] = v;
    }
  }
  __syncthreads();

  const int* imk = in_map  + (size_t)k * Mpairs;   // uniform bases
  const int* omk = out_map + (size_t)k * Mpairs;

  // --- A fragments: 16 rows x 128 channels, bf16, ISA 16-bit A layout ---
  // lane&15 = row within wave tile; lane half selects K sub-range (+0 / +8).
  const int rA   = m0 + wave * 16 + (lane & 15);
  const unsigned rAc = (unsigned)(rA < Mlast ? rA : Mlast);   // clamp
  const int gidx = imk[rAc];
  const int choff = (lane < 16) ? 0 : 8;
  const unsigned short* frow = featbf + (size_t)((unsigned)gidx << 7);

  v16bf A[4];
#pragma unroll
  for (int kk = 0; kk < 4; ++kk) {
    int cb = kk * 32 + choff;
    ((uint4*)&A[kk])[0] = *(const uint4*)(frow + cb);        // K +0..7
    ((uint4*)&A[kk])[1] = *(const uint4*)(frow + cb + 16);   // K +16..23
  }

  // --- per-lane 32-bit output element offsets (row*128 + col <= 12.8M) ---
  const int ncol = lane & 15;
  unsigned uoff[8];
#pragma unroll
  for (int i = 0; i < 8; ++i) {
    int mrow = (lane < 16) ? i : (8 + i);
    int r2 = m0 + wave * 16 + mrow;
    unsigned r2c = (unsigned)(r2 < Mlast ? r2 : Mlast);      // clamp
    int orow = omk[r2c];
    uoff[i] = ((unsigned)orow << 7) + (unsigned)ncol;
  }

  if (fullTile) {
    // ---- hot path: no masks anywhere ----
#pragma unroll
    for (int nt = 0; nt < 8; ++nt) {
      const unsigned short* wrow = &ldsW[(nt * 16 + ncol) * LDS_PITCH + choff];
      // Preload all 4 B fragments -> one ds clause, partial-wait drain.
      v16bf B[4];
#pragma unroll
      for (int kk = 0; kk < 4; ++kk) {
        int cb = kk * 32;
        ((uint4*)&B[kk])[0] = *(const uint4*)(wrow + cb);
        ((uint4*)&B[kk])[1] = *(const uint4*)(wrow + cb + 16);
      }
      v8f acc = {};
#pragma unroll
      for (int kk = 0; kk < 4; ++kk)
        acc = __builtin_amdgcn_wmma_f32_16x16x32_bf16(
            false, A[kk], false, B[kk], (short)0, acc, false, false);
#pragma unroll
      for (int i = 0; i < 8; ++i)
        unsafeAtomicAdd(out + (size_t)uoff[i] + nt * 16, acc[i]);
    }
  } else {
    // ---- edge path: branchless cndmask-to-zero, rows clamped to valid ----
    unsigned valid = 0u;
#pragma unroll
    for (int i = 0; i < 8; ++i) {
      int mrow = (lane < 16) ? i : (8 + i);
      if (m0 + wave * 16 + mrow < Mpairs) valid |= 1u << i;
    }
#pragma unroll
    for (int nt = 0; nt < 8; ++nt) {
      const unsigned short* wrow = &ldsW[(nt * 16 + ncol) * LDS_PITCH + choff];
      v16bf B[4];
#pragma unroll
      for (int kk = 0; kk < 4; ++kk) {
        int cb = kk * 32;
        ((uint4*)&B[kk])[0] = *(const uint4*)(wrow + cb);
        ((uint4*)&B[kk])[1] = *(const uint4*)(wrow + cb + 16);
      }
      v8f acc = {};
#pragma unroll
      for (int kk = 0; kk < 4; ++kk)
        acc = __builtin_amdgcn_wmma_f32_16x16x32_bf16(
            false, A[kk], false, B[kk], (short)0, acc, false, false);
#pragma unroll
      for (int i = 0; i < 8; ++i) {
        float val = (valid & (1u << i)) ? acc[i] : 0.0f;  // +0.0 add is inert
        unsafeAtomicAdd(out + (size_t)uoff[i] + nt * 16, val);
      }
    }
  }
}

extern "C" void kernel_launch(void* const* d_in, const int* in_sizes, int n_in,
                              void* d_out, int out_size, void* d_ws, size_t ws_size,
                              hipStream_t stream) {
  const float* features = (const float*)d_in[0];   // [N_in,128] fp32
  const float* kernelw  = (const float*)d_in[1];   // [K,128,128] fp32
  const float* bias     = (const float*)d_in[2];   // [1,128] fp32
  const int*   in_map   = (const int*)d_in[3];     // [K,M]
  const int*   out_map  = (const int*)d_in[4];     // [K,M]
  float* out = (float*)d_out;                      // [N_out,128] fp32

  const int kvol   = in_sizes[1] / (128 * 128);    // 27
  const int Mpairs = in_sizes[3] / kvol;           // 50000
  const int fElems = in_sizes[0];                  // N_in*128

  // Workspace layout: [bf16 W^T : kvol*16384 halves] [bf16 features]
  unsigned short* wt = (unsigned short*)d_ws;
  size_t wbytes = ((size_t)kvol * 16384 * 2 + 255) & ~(size_t)255;
  unsigned short* fb = (unsigned short*)((char*)d_ws + wbytes);

  int wtot = kvol * 16384;
  spc_conv_weights<<<(wtot + 255) / 256, 256, 0, stream>>>(kernelw, wt, wtot);

  int n4f = fElems / 4;
  spc_conv_feat<<<(n4f + 255) / 256, 256, 0, stream>>>(
      (const float4*)features, (uint2*)fb, n4f);

  int n4o = out_size / 4;
  spc_bias_init<<<(n4o + 255) / 256, 256, 0, stream>>>(
      (const float4*)bias, (float4*)out, n4o);

  dim3 grid((Mpairs + 127) / 128, kvol);
  spc_gemm<<<grid, 256, 0, stream>>>(fb, wt, in_map, out_map, out, Mpairs);
}
